// CoCov_65360812310717
// MI455X (gfx1250) — compile-verified, matrix operand order checked by reading, not run on hardware
//
#include <hip/hip_runtime.h>
#include <hip/hip_bf16.h>
#include <stdint.h>

// ---------------------------------------------------------------------------
// CoCov on MI455X (gfx1250): implicit-GEMM per 16x16 output tile.
//   M = 256 pixels, N = 64 out ch, K = 9 tile-offsets * 9 taps * 64 ch = 5184
// bf16 operands, f32 accumulation via v_wmma_f32_16x16x32_bf16 (wave32).
// Compute-bound (~348 GFLOP vs <1 GB of traffic), so everything is organized
// around feeding the WMMA pipes from LDS.
// ---------------------------------------------------------------------------

typedef __attribute__((ext_vector_type(8)))  __bf16 v8bf;
typedef __attribute__((ext_vector_type(16))) __bf16 v16bf;
typedef __attribute__((ext_vector_type(8)))  float  v8f;
typedef __attribute__((ext_vector_type(4)))  int    v4i;

// Address-space-qualified pointer types for the async-LDS builtin
// (param 0 is v4i addrspace(1)*, param 1 is v4i addrspace(3)* per clang).
typedef __attribute__((address_space(1))) v4i* gas_v4i_ptr;
typedef __attribute__((address_space(3))) v4i* las_v4i_ptr;

#if __has_builtin(__builtin_amdgcn_global_load_async_to_lds_b128)
#define HAVE_ASYNC_LDS 1
#else
#define HAVE_ASYNC_LDS 0
#endif

// 16-byte global -> LDS copy. Async (ASYNCcnt-tracked) when available.
__device__ __forceinline__ void copy16(const __bf16* g, __bf16* l) {
#if HAVE_ASYNC_LDS
  __builtin_amdgcn_global_load_async_to_lds_b128(
      (gas_v4i_ptr)(void*)(g),
      (las_v4i_ptr)(void*)(l),
      0, 0);
#else
  *(uint4*)l = *(const uint4*)g;
#endif
}

__device__ __forceinline__ void wait_async0() {
#if HAVE_ASYNC_LDS
#if __has_builtin(__builtin_amdgcn_s_wait_asynccnt)
  __builtin_amdgcn_s_wait_asynccnt(0);
#else
  asm volatile("s_wait_asynccnt 0" ::: "memory");
#endif
#endif
}

// Load one 16x32 bf16 fragment row-slice for this lane: two aligned 16B LDS
// loads (elems 0..7 at +0, elems 8..15 at +16 halfs), per CDNA5 A layout.
__device__ __forceinline__ v16bf ldfrag(const __bf16* p) {
  v8bf lo = *(const v8bf*)(p);
  v8bf hi = *(const v8bf*)(p + 16);
  return __builtin_shufflevector(lo, hi, 0, 1, 2, 3, 4, 5, 6, 7,
                                         8, 9, 10, 11, 12, 13, 14, 15);
}

// ---------------- pre-pass 1: input f32 NCHW -> bf16 tile-major, c-last ----
// Xt[b][u1][u2][v1*16+v2][c]   (each 16x16x64 tile = contiguous 32KB)
__global__ void prep_x(const float* __restrict__ x, __bf16* __restrict__ Xt) {
  int pid = blockIdx.x * 256 + threadIdx.x;      // 8*256*256 threads
  int b = pid >> 16;
  int H = (pid >> 8) & 255;
  int W = pid & 255;
  int u1 = H >> 4, v1 = H & 15, u2 = W >> 4, v2 = W & 15;
  const float* src = x + ((size_t)(b * 64) * 256 + H) * 256 + W;
  __align__(16) __bf16 tmp[64];
#pragma unroll
  for (int c = 0; c < 64; ++c) tmp[c] = (__bf16)src[(size_t)c * 65536];
  __bf16* dst =
      Xt + ((((size_t)b * 16 + u1) * 16 + u2) * 256 + (v1 * 16 + v2)) * 64;
#pragma unroll
  for (int j = 0; j < 8; ++j) ((uint4*)dst)[j] = ((const uint4*)tmp)[j];
}

// ---------------- pre-pass 2: weights -> bf16 Wt[k][dv1][dv2][o][72] -------
// (row padded 64->72 halfs so LDS fragment reads spread across banks)
__global__ void prep_w(const float* __restrict__ kern, __bf16* __restrict__ Wt) {
  int idx = blockIdx.x * 256 + threadIdx.x;
  if (idx >= 9 * 3 * 3 * 64 * 72) return;
  int cpad = idx % 72;
  int t = idx / 72;
  int o = t % 64; t /= 64;
  int dv2 = t % 3; t /= 3;
  int dv1 = t % 3;
  int k = t / 3;
  float v = 0.f;
  if (cpad < 64)
    v = kern[((((size_t)k * 64 + o) * 64 + cpad) * 3 + dv1) * 3 + dv2];
  Wt[idx] = (__bf16)v;
}

// ---------------- main kernel ----------------------------------------------
#define XS_HALFS (18 * 18 * 72) /* 23328: padded tile with zero halo ring */
#define WS_HALFS (3 * 3 * 64 * 72) /* 41472: one tile-offset's weights    */
#define LDS_BYTES ((XS_HALFS + WS_HALFS) * 2) /* 129600 B -> 2 WG / WGP   */

__global__ __launch_bounds__(256) void cocov_main(
    const __bf16* __restrict__ Xt, const __bf16* __restrict__ Wt,
    float* __restrict__ out) {
  extern __shared__ __bf16 smem[];
  __bf16* Xs = smem;             // [v1p(18)][v2p(18)][c(72)]
  __bf16* Ws = smem + XS_HALFS;  // [dv1][dv2][o(64)][c(72)]

  const int tid = threadIdx.x;
  const int lane = tid & 31;
  const int wave = tid >> 5;     // 8 waves
  const int lhalf = lane & 15;
  const int hig = lane >> 4;     // lane-group select (K-halves / M+8)

  const int blk = blockIdx.x;    // 2048 = 8 * 16 * 16
  const int b = blk >> 8;
  const int U = (blk >> 4) & 15;
  const int U2 = blk & 15;

  v8f acc[2][4] = {};            // wave owns M rows [2w,2w+2) x all 4 N tiles

  // Zero the whole Xs (establishes the zero halo ring once).
  {
    uint4 z = {0u, 0u, 0u, 0u};
    for (int i = tid; i < XS_HALFS / 8; i += 256) ((uint4*)Xs)[i] = z;
  }

#pragma unroll 1
  for (int k = 0; k < 9; ++k) {
    const int u1s = U + (k / 3) - 1;
    const int u2s = U2 + (k % 3) - 1;
    // Shifted-out tiles are exactly zero (16-pixel outer pad) -> skip.
    if ((unsigned)u1s >= 16u || (unsigned)u2s >= 16u) continue;  // uniform

    __syncthreads();  // previous iteration's reads done before overwrite

    // Stage this tile's 16x16x64 bf16 block into the LDS interior.
    {
      const int v1 = tid >> 4, v2 = tid & 15;
      const __bf16* src =
          Xt + ((((size_t)b * 16 + u1s) * 16 + u2s) * 256 + tid) * 64;
      __bf16* dst = Xs + ((v1 + 1) * 18 + (v2 + 1)) * 72;
#pragma unroll
      for (int j = 0; j < 8; ++j) copy16(src + j * 8, dst + j * 8);
    }
    // Stage this tile-offset's weight block (3*3*64*72 halfs).
    {
      const __bf16* wsrc = Wt + (size_t)k * WS_HALFS;
      for (int i = tid; i < WS_HALFS / 8; i += 256)
        copy16(wsrc + i * 8, Ws + i * 8);
    }
    wait_async0();
    __syncthreads();

    // 3x3 taps x 2 channel-chunks of K=32 -> 18 steps x 8 WMMAs.
#pragma unroll
    for (int dv1 = 0; dv1 < 3; ++dv1)
#pragma unroll
      for (int dv2 = 0; dv2 < 3; ++dv2)
#pragma unroll
        for (int cc = 0; cc < 2; ++cc) {
          const int cbase = cc * 32 + hig * 8;
          v16bf a[2], bb[4];
#pragma unroll
          for (int i = 0; i < 2; ++i) {
            const int mt = wave * 2 + i;  // V1 row of this M subtile
            a[i] = ldfrag(Xs + ((mt + dv1) * 18 + (lhalf + dv2)) * 72 + cbase);
          }
#pragma unroll
          for (int j = 0; j < 4; ++j)
            bb[j] = ldfrag(Ws + ((dv1 * 3 + dv2) * 64 + j * 16 + lhalf) * 72 +
                           cbase);
#pragma unroll
          for (int i = 0; i < 2; ++i)
#pragma unroll
            for (int j = 0; j < 4; ++j)
              acc[i][j] = __builtin_amdgcn_wmma_f32_16x16x32_bf16(
                  false, a[i], false, bb[j], (short)0, acc[i][j], false, false);
        }
  }

  // Store: D element v of lane -> V2 = v + 8*hig, o = j*16 + lhalf, V1 = mt.
#pragma unroll
  for (int i = 0; i < 2; ++i) {
    const int V1 = wave * 2 + i;
#pragma unroll
    for (int j = 0; j < 4; ++j) {
      const int o = j * 16 + lhalf;
      size_t base = (((size_t)(b * 64 + o) * 256) + (U * 16 + V1)) * 256 +
                    U2 * 16 + hig * 8;
      float4 f0 = make_float4(acc[i][j][0], acc[i][j][1], acc[i][j][2],
                              acc[i][j][3]);
      float4 f1 = make_float4(acc[i][j][4], acc[i][j][5], acc[i][j][6],
                              acc[i][j][7]);
      *(float4*)(out + base) = f0;
      *(float4*)(out + base + 4) = f1;
    }
  }
}

// ---------------- launcher --------------------------------------------------
extern "C" void kernel_launch(void* const* d_in, const int* in_sizes, int n_in,
                              void* d_out, int out_size, void* d_ws,
                              size_t ws_size, hipStream_t stream) {
  const float* x = (const float*)d_in[0];     // (8,64,256,256) f32
  const float* kern = (const float*)d_in[1];  // (9,64,64,3,3) f32
  float* out = (float*)d_out;                 // (8,64,256,256) f32

  // Workspace: Xt (64 MiB bf16) + Wt (0.71 MiB bf16) = ~64.7 MiB.
  __bf16* Xt = (__bf16*)d_ws;
  __bf16* Wt = Xt + (size_t)8 * 16 * 16 * 256 * 64;

  prep_x<<<2048, 256, 0, stream>>>(x, Xt);
  prep_w<<<1458, 256, 0, stream>>>(kern, Wt);
  cocov_main<<<2048, 256, LDS_BYTES, stream>>>(Xt, Wt, out);
}
